// DN4PlusPlus_50740743635203
// MI455X (gfx1250) — matrix operand order repaired and checked by reading, not compiled, not run on hardware
//
#include <hip/hip_runtime.h>
#include <hip/hip_bf16.h>

// ---------------------------------------------------------------------------
// DN4++ episode head for MI455X (gfx1250, wave32, WMMA).
//
// Roofline: ~11 GFLOP total, dominated by two 4096x8192x64 GEMMs. Inputs are
// ~24MB (=> ~1us at 23.3 TB/s HBM), so the only way to lose is materializing
// the 128MB sim matrix. We fuse sim -> softmax -> PV flash-style in registers
// on v_wmma_f32_16x16x32_f16. Because sim is a cosine in [-1,1], TEMP*sim is
// bounded by +-10, so softmax needs NO running max: exp(10*sim) <= 2.2e4 and
// the 8192-term sum fits f32 easily. That removes all per-chunk shfl trees /
// rescales from the hot loop. sn chunks are staged via the Tensor Data Mover
// (tensor_load_to_lds + s_wait_tensorcnt); this toolchain uses the 6-arg
// builtin form (g0,g1,g2,g3,g4,cpol).
// ---------------------------------------------------------------------------

typedef __attribute__((ext_vector_type(16))) _Float16 v16h;
typedef __attribute__((ext_vector_type(8)))  float    v8f;
typedef __attribute__((ext_vector_type(4)))  unsigned int v4u;
typedef __attribute__((ext_vector_type(8)))  int      v8i_;
typedef __attribute__((ext_vector_type(4)))  int      v4i_;

#define N_Q     4096
#define N_S     8192
#define D_IN    512
#define D_H     128
#define D_H2    64
#define D_H4    32
#define TEMP_F  10.0f
#define BN_EPS  1e-5f
#define COS_EPS 1e-8f

struct DnParams {
  const float *x, *sup;
  const float *W1, *b1, *g1, *bb1, *m1, *v1;
  const float *W2, *b2, *g2, *bb2, *m2, *v2;
  const float *W3, *b3, *g3, *bb3, *m3, *v3;
  const float *Wa1, *ba1, *Wa2, *ba2;
  const float *Wp1, *bp1, *gp, *bbp, *mp, *vp, *Wp2, *bp2;
  _Float16 *qn, *sn, *ws;   // scratch in d_ws
  float *out;
};

// --- WMMA fragment index helpers (CDNA5 ISA 7.12.2, wave32) -----------------
__device__ __forceinline__ int a_k(int i, int lane) {
  int g = i >> 1, j = i & 1;
  return 2 * (g & 3) + j + ((g >= 4) ? 16 : 0) + ((lane >= 16) ? 8 : 0);
}

__device__ __forceinline__ v16h load_a_f32(const float* base, int ld, int k0, int lane) {
  const float* p = base + (size_t)(lane & 15) * ld + k0;
  v16h a;
#pragma unroll
  for (int i = 0; i < 16; ++i) a[i] = (_Float16)p[a_k(i, lane)];
  return a;
}

__device__ __forceinline__ v16h load_a_f16(const _Float16* base, int ld, int k0, int lane) {
  const _Float16* p = base + (size_t)(lane & 15) * ld + k0;
  v16h a;
#pragma unroll
  for (int i = 0; i < 16; ++i) a[i] = p[a_k(i, lane)];
  return a;
}

// B 32x16 f16 fragment from a [n][k] layout with row stride ld:
// lane holds column n = n0 + (lane&15); K = a_k(i, lane) (same order as A).
__device__ __forceinline__ v16h load_b_nk(const _Float16* base, int ld, int n0, int k0, int lane) {
  const _Float16* p = base + (size_t)(n0 + (lane & 15)) * ld + k0;
  v16h b;
#pragma unroll
  for (int i = 0; i < 16; ++i) b[i] = p[a_k(i, lane)];
  return b;
}

__device__ __forceinline__ v8f wmma16(v16h a, v16h b, v8f c) {
  return __builtin_amdgcn_wmma_f32_16x16x32_f16(false, a, false, b, (short)0, c,
                                                false, false);
}

// --- Tensor Data Mover: 1-D contiguous f16 copy global -> LDS (ISA ch.8) ---
// D# group0: count=1 | lds_addr | global_addr[56:0] | type=2.
// D# group1: data_size=2B; tensor_dim0=tile_dim0=nelem; dim1=1; stride=nelem.
#if defined(__has_builtin)
#if __has_builtin(__builtin_amdgcn_tensor_load_to_lds) && __has_builtin(__builtin_amdgcn_s_wait_tensorcnt)
#define HAVE_TDM 1
#endif
#endif

#ifdef HAVE_TDM
__device__ __forceinline__ void tdm_copy_f16(unsigned lds_addr, const _Float16* gptr,
                                             unsigned nelem) {
  unsigned long long ga = (unsigned long long)(uintptr_t)gptr;
  v4u g0;
  g0[0] = 1u;                                        // count=1, user mode
  g0[1] = lds_addr;                                  // bits[63:32]
  g0[2] = (unsigned)(ga & 0xffffffffu);              // global_addr[31:0]
  g0[3] = (unsigned)((ga >> 32) & 0x01ffffffu) | (2u << 30);  // [56:32] | type=2
  v8i_ g1;
  g1[0] = (1 << 16);                                 // data_size=1 -> 2 bytes
  g1[1] = (int)((nelem & 0xffffu) << 16);            // tensor_dim0[15:0] @63:48
  g1[2] = (int)((nelem >> 16) & 0xffffu) | (1 << 16);// dim0[31:16] | tensor_dim1=1
  g1[3] = (int)((nelem & 0xffffu) << 16);            // tile_dim0 @127:112
  g1[4] = 0;                                         // tile_dim1/2 unused
  g1[5] = (int)nelem;                                // tensor_dim0_stride lo
  g1[6] = 0;
  g1[7] = 0;
  v4i_ z4 = {0, 0, 0, 0};
  v8i_ z8 = {0, 0, 0, 0, 0, 0, 0, 0};
  // 6-arg toolchain form: (g0, g1, g2, g3, g4, cpol)
  __builtin_amdgcn_tensor_load_to_lds(g0, g1, z4, z4, z8, 0);
}
#endif

// ---------------------------------------------------------------------------
// Kernel 1: fused descriptor MLP for all 12288 rows.
// Block = 256 threads = 8 waves; wave owns a 16-row tile. 96 blocks.
// ---------------------------------------------------------------------------
__global__ __launch_bounds__(256) void dn4_desc_kernel(DnParams P) {
  const int lane = threadIdx.x & 31;
  const int wave = threadIdx.x >> 5;
  const int l15  = lane & 15;
  const int row0 = blockIdx.x * 128 + wave * 16;      // [0, 12288)
  const bool isQ = row0 < N_Q;
  const float* src = isQ ? P.x : P.sup;
  const size_t srcRow0 = (size_t)(isQ ? row0 : row0 - N_Q);

  __shared__ _Float16 sBT[D_H * 32];        // weight slab, [n][k], 8KB
  __shared__ _Float16 sH[8][16 * D_H];      // per-wave activation tile, 32KB
  __shared__ float    sA[8][16 * 33 + 16];  // attention scratch, ~17KB

  // ---- layer 1: h1 = bn1(relu(x @ W1 + b1)), 512 -> 128 --------------------
  v8f acc1[8];
#pragma unroll
  for (int nt = 0; nt < 8; ++nt) { v8f z = {}; acc1[nt] = z; }
  for (int ks = 0; ks < 16; ++ks) {
    __syncthreads();
    for (int t = threadIdx.x; t < D_H * 32; t += 256) {
      int n = t >> 5, kk = t & 31;
      sBT[n * 32 + kk] = (_Float16)P.W1[(size_t)(32 * ks + kk) * D_H + n];
    }
    __syncthreads();
    v16h a = load_a_f32(src + srcRow0 * D_IN, D_IN, 32 * ks, lane);
#pragma unroll
    for (int nt = 0; nt < 8; ++nt)
      acc1[nt] = wmma16(a, load_b_nk(sBT, 32, nt * 16, 0, lane), acc1[nt]);
  }
#pragma unroll
  for (int nt = 0; nt < 8; ++nt) {
    int c = nt * 16 + l15;
    float sc = P.g1[c] / sqrtf(P.v1[c] + BN_EPS);
    float bi = P.b1[c], mm = P.m1[c], bb = P.bb1[c];
#pragma unroll
    for (int r = 0; r < 8; ++r) {
      int m = r + ((lane < 16) ? 0 : 8);
      float y = fmaxf(acc1[nt][r] + bi, 0.0f);
      sH[wave][m * D_H + c] = (_Float16)((y - mm) * sc + bb);
    }
  }

  // ---- layer 2: h2 = bn2(relu(h1 @ W2 + b2)), 128 -> 128 -------------------
  v8f acc2[8];
#pragma unroll
  for (int nt = 0; nt < 8; ++nt) { v8f z = {}; acc2[nt] = z; }
  for (int ks = 0; ks < 4; ++ks) {
    __syncthreads();
    for (int t = threadIdx.x; t < D_H * 32; t += 256) {
      int n = t >> 5, kk = t & 31;
      sBT[n * 32 + kk] = (_Float16)P.W2[(size_t)(32 * ks + kk) * D_H + n];
    }
    __syncthreads();
    v16h a = load_a_f16(&sH[wave][0], D_H, 32 * ks, lane);
#pragma unroll
    for (int nt = 0; nt < 8; ++nt)
      acc2[nt] = wmma16(a, load_b_nk(sBT, 32, nt * 16, 0, lane), acc2[nt]);
  }
#pragma unroll
  for (int nt = 0; nt < 8; ++nt) {
    int c = nt * 16 + l15;
    float sc = P.g2[c] / sqrtf(P.v2[c] + BN_EPS);
    float bi = P.b2[c], mm = P.m2[c], bb = P.bb2[c];
#pragma unroll
    for (int r = 0; r < 8; ++r) {
      int m = r + ((lane < 16) ? 0 : 8);
      float y = fmaxf(acc2[nt][r] + bi, 0.0f);
      sH[wave][m * D_H + c] = (_Float16)((y - mm) * sc + bb);
    }
  }

  // ---- layer 3: d = bn3(relu(h2 @ W3 + b3)), 128 -> 64 ---------------------
  v8f acc3[4];
#pragma unroll
  for (int nt = 0; nt < 4; ++nt) { v8f z = {}; acc3[nt] = z; }
  for (int ks = 0; ks < 4; ++ks) {
    __syncthreads();
    for (int t = threadIdx.x; t < D_H2 * 32; t += 256) {
      int n = t >> 5, kk = t & 31;
      sBT[n * 32 + kk] = (_Float16)P.W3[(size_t)(32 * ks + kk) * D_H2 + n];
    }
    __syncthreads();
    v16h a = load_a_f16(&sH[wave][0], D_H, 32 * ks, lane);
#pragma unroll
    for (int nt = 0; nt < 4; ++nt)
      acc3[nt] = wmma16(a, load_b_nk(sBT, 32, nt * 16, 0, lane), acc3[nt]);
  }
#pragma unroll
  for (int nt = 0; nt < 4; ++nt) {
    int c = nt * 16 + l15;
    float sc = P.g3[c] / sqrtf(P.v3[c] + BN_EPS);
    float bi = P.b3[c], mm = P.m3[c], bb = P.bb3[c];
#pragma unroll
    for (int r = 0; r < 8; ++r) {
      int m = r + ((lane < 16) ? 0 : 8);
      float y = fmaxf(acc3[nt][r] + bi, 0.0f);
      y = (y - mm) * sc + bb;
      acc3[nt][r] = y;                              // keep f32 d in registers
      sH[wave][m * D_H2 + c] = (_Float16)y;         // f16 copy for attn A-op
    }
  }

  // ---- attention gate: sigmoid(relu(d@Wa1+ba1)@Wa2+ba2) --------------------
  v8f acca[2];
#pragma unroll
  for (int nt = 0; nt < 2; ++nt) { v8f z = {}; acca[nt] = z; }
  for (int ks = 0; ks < 2; ++ks) {
    __syncthreads();
    for (int t = threadIdx.x; t < D_H4 * 32; t += 256) {
      int n = t >> 5, kk = t & 31;
      sBT[n * 32 + kk] = (_Float16)P.Wa1[(size_t)(32 * ks + kk) * D_H4 + n];
    }
    __syncthreads();
    v16h a = load_a_f16(&sH[wave][0], D_H2, 32 * ks, lane);
#pragma unroll
    for (int nt = 0; nt < 2; ++nt)
      acca[nt] = wmma16(a, load_b_nk(sBT, 32, nt * 16, 0, lane), acca[nt]);
  }
#pragma unroll
  for (int nt = 0; nt < 2; ++nt) {
    int c = nt * 16 + l15;
    float bi = P.ba1[c];
#pragma unroll
    for (int r = 0; r < 8; ++r) {
      int m = r + ((lane < 16) ? 0 : 8);
      sA[wave][m * 33 + c] = fmaxf(acca[nt][r] + bi, 0.0f);
    }
  }
  if (lane < 16) {
    float logit = P.ba2[0];
#pragma unroll
    for (int j = 0; j < D_H4; ++j) logit += sA[wave][lane * 33 + j] * P.Wa2[j];
    sA[wave][16 * 33 + lane] = 1.0f / (1.0f + __expf(-logit));
  }

  // ---- wd = d*att, row L2 norm, emit qn/sn/ws ------------------------------
#pragma unroll
  for (int r = 0; r < 8; ++r) {
    int m = r + ((lane < 16) ? 0 : 8);
    float att = sA[wave][16 * 33 + m];
    float sq = 0.0f;
#pragma unroll
    for (int nt = 0; nt < 4; ++nt) {
      float w = acc3[nt][r] * att;
      acc3[nt][r] = w;
      sq += w * w;
    }
#pragma unroll
    for (int off = 1; off < 16; off <<= 1) sq += __shfl_xor(sq, off, 16);
    float inv = 1.0f / fmaxf(sqrtf(sq), COS_EPS);
    int grow = row0 + m;
#pragma unroll
    for (int nt = 0; nt < 4; ++nt) {
      int c = nt * 16 + l15;
      float w = acc3[nt][r];
      if (isQ) {
        P.qn[(size_t)grow * D_H2 + c] = (_Float16)(w * inv);
      } else {
        size_t sr = (size_t)(grow - N_Q);
        P.sn[sr * D_H2 + c] = (_Float16)(w * inv);
        P.ws[sr * D_H2 + c] = (_Float16)w;
      }
    }
  }
}

// ---------------------------------------------------------------------------
// Kernel 2: fused sim -> bounded softmax -> feat -> predictor.
// Block = 8 waves x 16 queries; 32 blocks. S chunk = 64 (128 iterations).
// 16 WMMAs per chunk; softmax is pure per-lane exp/add (max fixed at 0, valid
// since |TEMP*sim| <= 10); denominators reduced once at the end. sn chunk
// staged via TDM; ws chunk transposed manually. 32KB LDS total (phases alias).
// ---------------------------------------------------------------------------
__global__ __launch_bounds__(256) void dn4_attn_kernel(DnParams P) {
  const int lane = threadIdx.x & 31;
  const int wave = threadIdx.x >> 5;
  const int l15  = lane & 15;
  const int q0   = blockIdx.x * 128 + wave * 16;

  __shared__ char smem[32768];
  _Float16* snC = (_Float16*)smem;                   // [64 s][64 d], 8KB
  _Float16* wsT = (_Float16*)(smem + 8192);          // [64 d][64 s], 8KB
  _Float16* sP  = (_Float16*)(smem + 16384) + wave * (16 * 64);  // P tile, 2KB/wave

  v16h aq0 = load_a_f16(P.qn + (size_t)q0 * D_H2, D_H2, 0, lane);
  v16h aq1 = load_a_f16(P.qn + (size_t)q0 * D_H2, D_H2, 32, lane);

  v8f acc[4];
#pragma unroll
  for (int nt = 0; nt < 4; ++nt) { v8f z = {}; acc[nt] = z; }
  float lsum[8];
#pragma unroll
  for (int r = 0; r < 8; ++r) lsum[r] = 0.0f;

  for (int s0 = 0; s0 < N_S; s0 += 64) {
    __syncthreads();                 // previous chunk fully consumed
#ifdef HAVE_TDM
    if (wave == 0)
      tdm_copy_f16((unsigned)(uintptr_t)snC, P.sn + (size_t)s0 * D_H2, 64 * D_H2);
#else
    for (int t = threadIdx.x; t < 64 * D_H2; t += 256)
      snC[t] = P.sn[(size_t)s0 * D_H2 + t];
#endif
    for (int t = threadIdx.x; t < 64 * D_H2; t += 256) {
      int s = t >> 6, d = t & 63;
      wsT[d * 64 + s] = P.ws[(size_t)(s0 + s) * D_H2 + d];
    }
    if (s0 + 64 < N_S) __builtin_prefetch(P.ws + (size_t)(s0 + 64) * D_H2, 0, 1);
#ifdef HAVE_TDM
    if (wave == 0) __builtin_amdgcn_s_wait_tensorcnt(0);
#endif
    __syncthreads();                 // chunk published

    // sim tile 16x64: 4 n-tiles x 2 K-frags = 8 WMMAs
    v8f c[4];
#pragma unroll
    for (int nt = 0; nt < 4; ++nt) {
      v8f z = {}; c[nt] = z;
      c[nt] = wmma16(aq0, load_b_nk(snC, D_H2, nt * 16, 0, lane), c[nt]);
      c[nt] = wmma16(aq1, load_b_nk(snC, D_H2, nt * 16, 32, lane), c[nt]);
    }

    // bounded softmax numerators: p = exp(TEMP*sim), no max/rescale needed
#pragma unroll
    for (int r = 0; r < 8; ++r) {
      int m = r + ((lane < 16) ? 0 : 8);
#pragma unroll
      for (int nt = 0; nt < 4; ++nt) {
        float p = __expf(c[nt][r] * TEMP_F);
        lsum[r] += p;
        sP[m * 64 + nt * 16 + l15] = (_Float16)p;
      }
    }

    // feat += P (16x64) @ ws_chunk (64x64): 4 n-tiles x 2 K-frags = 8 WMMAs
    v16h ap0 = load_a_f16(sP, 64, 0, lane);
    v16h ap1 = load_a_f16(sP, 64, 32, lane);
#pragma unroll
    for (int nt = 0; nt < 4; ++nt) {
      acc[nt] = wmma16(ap0, load_b_nk(wsT, 64, nt * 16, 0, lane), acc[nt]);
      acc[nt] = wmma16(ap1, load_b_nk(wsT, 64, nt * 16, 32, lane), acc[nt]);
    }
  }

  // ---- finalize: reduce denominators once, park feat, run predictor -------
  __syncthreads();                   // all waves done with snC/wsT/sP
  float* sF = (float*)smem;          // alias: 8 waves x 16x64 f32 = 32KB
#pragma unroll
  for (int r = 0; r < 8; ++r) {
    float s = lsum[r];
#pragma unroll
    for (int off = 1; off < 16; off <<= 1) s += __shfl_xor(s, off, 16);
    float linv = 1.0f / fmaxf(s, 1e-30f);
    int m = r + ((lane < 16) ? 0 : 8);
#pragma unroll
    for (int nt = 0; nt < 4; ++nt)
      sF[wave * 1024 + m * D_H2 + nt * 16 + l15] = acc[nt][r] * linv;
  }

  if (lane < 16) {   // predictor: bn(relu(feat@Wp1+bp1)) @ Wp2 + bp2
    float o = P.bp2[0];
    for (int j = 0; j < D_H4; ++j) {
      float h = P.bp1[j];
#pragma unroll
      for (int k = 0; k < D_H2; ++k)
        h += sF[wave * 1024 + lane * D_H2 + k] * P.Wp1[k * D_H4 + j];
      h = fmaxf(h, 0.0f);
      h = (h - P.mp[j]) * (P.gp[j] / sqrtf(P.vp[j] + BN_EPS)) + P.bbp[j];
      o += h * P.Wp2[j];
    }
    P.out[q0 + lane] = o;
  }
}

// ---------------------------------------------------------------------------
// Host launch. Param order assumes jax pytree flattening of the params dict:
// top-level keys sorted (capitals first): W1,W2,W3,Wa1,Wa2,Wp1,Wp2,b1,b2,b3,
// ba1,ba2,bn1,bn2,bn3,bnp,bp1,bp2 ; each bn dict sorted: b,g,m,v.
// ---------------------------------------------------------------------------
extern "C" void kernel_launch(void* const* d_in, const int* in_sizes, int n_in,
                              void* d_out, int out_size, void* d_ws, size_t ws_size,
                              hipStream_t stream) {
  (void)in_sizes; (void)n_in; (void)out_size; (void)ws_size;
  DnParams P;
  P.x   = (const float*)d_in[0];
  P.sup = (const float*)d_in[1];
  P.W1  = (const float*)d_in[2];
  P.W2  = (const float*)d_in[3];
  P.W3  = (const float*)d_in[4];
  P.Wa1 = (const float*)d_in[5];
  P.Wa2 = (const float*)d_in[6];
  P.Wp1 = (const float*)d_in[7];
  P.Wp2 = (const float*)d_in[8];
  P.b1  = (const float*)d_in[9];
  P.b2  = (const float*)d_in[10];
  P.b3  = (const float*)d_in[11];
  P.ba1 = (const float*)d_in[12];
  P.ba2 = (const float*)d_in[13];
  P.bb1 = (const float*)d_in[14]; P.g1 = (const float*)d_in[15];
  P.m1  = (const float*)d_in[16]; P.v1 = (const float*)d_in[17];
  P.bb2 = (const float*)d_in[18]; P.g2 = (const float*)d_in[19];
  P.m2  = (const float*)d_in[20]; P.v2 = (const float*)d_in[21];
  P.bb3 = (const float*)d_in[22]; P.g3 = (const float*)d_in[23];
  P.m3  = (const float*)d_in[24]; P.v3 = (const float*)d_in[25];
  P.bbp = (const float*)d_in[26]; P.gp = (const float*)d_in[27];
  P.mp  = (const float*)d_in[28]; P.vp = (const float*)d_in[29];
  P.bp1 = (const float*)d_in[30];
  P.bp2 = (const float*)d_in[31];

  char* ws = (char*)d_ws;
  P.qn = (_Float16*)ws;                                  // 4096*64 f16
  P.sn = (_Float16*)(ws + (size_t)N_Q * D_H2 * 2);       // 8192*64 f16
  P.ws = (_Float16*)(ws + (size_t)(N_Q + N_S) * D_H2 * 2);
  P.out = (float*)d_out;

  dn4_desc_kernel<<<(N_Q + N_S) / 128, 256, 0, stream>>>(P);
  dn4_attn_kernel<<<N_Q / 128, 256, 0, stream>>>(P);
}
